// IntraRNN_37460704755938
// MI455X (gfx1250) — matrix operand order repaired and checked by reading, not compiled, not run on hardware
//
#include <hip/hip_runtime.h>
#include <hip/hip_bf16.h>

// ---------------------------------------------------------------------------
// Problem constants (B, S, H, V) = (256, 50, 512, 100000)
// ---------------------------------------------------------------------------
constexpr int Bc = 256;
constexpr int Sc = 50;
constexpr int Hc = 512;
constexpr int Vc = 100000;

typedef __bf16 bf16_t;
typedef bf16_t v16bf __attribute__((ext_vector_type(16)));
typedef bf16_t v8bf  __attribute__((ext_vector_type(8)));
typedef float  v8f   __attribute__((ext_vector_type(8)));

#define WMMA_BF16(a, b, c) \
    __builtin_amdgcn_wmma_f32_16x16x32_bf16(false, (a), false, (b), (short)0, (c), false, false)

__device__ __forceinline__ v8f zero_v8f() {
    v8f z;
#pragma unroll
    for (int i = 0; i < 8; ++i) z[i] = 0.0f;
    return z;
}

__device__ __forceinline__ v8bf cvt8(float4 a, float4 b) {
    v8bf r;
    r[0] = (bf16_t)a.x; r[1] = (bf16_t)a.y; r[2] = (bf16_t)a.z; r[3] = (bf16_t)a.w;
    r[4] = (bf16_t)b.x; r[5] = (bf16_t)b.y; r[6] = (bf16_t)b.z; r[7] = (bf16_t)b.w;
    return r;
}

__device__ __forceinline__ v16bf cat16(v8bf lo, v8bf hi) {
    return __builtin_shufflevector(lo, hi, 0, 1, 2, 3, 4, 5, 6, 7,
                                           8, 9, 10, 11, 12, 13, 14, 15);
}

// A fragment (16x32, bf16) for v_wmma_f32_16x16x32_bf16, loaded from fp32
// row-major A, rows m0..m0+15, K window [k0, k0+32).
// Layout (ISA 7.12.2): lane L, m = L%16, half = L/16;
//   elements e<8 : K = half*8 + e        -> 16 contiguous bytes (fp32: 32B)
//   elements e>=8: K = 16 + half*8 + e-8 -> 16 contiguous bytes at +16 elems
__device__ __forceinline__ v16bf load_a_f32(const float* __restrict__ A, long lda,
                                            int m0, int k0, int lane) {
    int m = m0 + (lane & 15);
    int half = lane >> 4;
    const float4* q = (const float4*)(A + (long)m * lda + k0 + half * 8);
    v8bf lo = cvt8(q[0], q[1]);  // K = half*8 + 0..7
    v8bf hi = cvt8(q[4], q[5]);  // K = 16 + half*8 + 0..7
    return cat16(lo, hi);
}

// B fragment (32x16, bf16): B[k,n] taken from row-major weight W(N,K),
// i.e. B[k,n] = W[n*ldb + k]. Layout: lane L, n = L%16, khalf = L/16,
// element e<8: K = khalf*8 + e; e>=8: K = 16 + khalf*8 + e-8.
__device__ __forceinline__ v16bf load_b_f32(const float* __restrict__ W, long ldb,
                                            int n0, int k0, int lane) {
    int n = n0 + (lane & 15);
    int kh = lane >> 4;
    const float4* q = (const float4*)(W + (long)n * ldb + k0 + kh * 8);
    v8bf lo = cvt8(q[0], q[1]);  // K = kh*8 + 0..7
    v8bf hi = cvt8(q[4], q[5]);  // K = 16 + kh*8 + 0..7
    return cat16(lo, hi);
}

// ---------------------------------------------------------------------------
// Generic GEMM with bias: C(M,N) = A(M,K) @ W(N,K)^T + bias(N)
// One wave per 16x16 output tile. Used for hid_proj, gi, gh.
// ---------------------------------------------------------------------------
__global__ void gemm_bias_kernel(const float* __restrict__ A, long lda,
                                 const float* __restrict__ W, long ldb,
                                 const float* __restrict__ bias,
                                 float* __restrict__ C, long ldc,
                                 int Mtiles, int Ntiles, int K) {
    int lane = threadIdx.x & 31;
    int gw = blockIdx.x * (blockDim.x >> 5) + (threadIdx.x >> 5);
    if (gw >= Mtiles * Ntiles) return;
    int nt = gw % Ntiles, mt = gw / Ntiles;
    int m0 = mt * 16, n0 = nt * 16;

    v8f acc = zero_v8f();
    for (int k0 = 0; k0 < K; k0 += 32) {
        v16bf a = load_a_f32(A, lda, m0, k0, lane);
        v16bf b = load_b_f32(W, ldb, n0, k0, lane);
        acc = WMMA_BF16(a, b, acc);
    }
    // C/D layout: N = lane%16, M = r + 8*(lane/16)
    int n = n0 + (lane & 15);
    int half = lane >> 4;
    float bv = bias[n];
#pragma unroll
    for (int r = 0; r < 8; ++r)
        C[(long)(m0 + r + 8 * half) * ldc + n] = acc[r] + bv;
}

// ---------------------------------------------------------------------------
// Attention scores: scores[b,s] = sum_n v[n] * tanh( inter[b,s,:]@W_attn[n,:H]
//                                                    + hid_proj[b,n] )
// hid_proj already contains h@W_attn[:,H:].T + b_attn.
// One wave per 16-row tile of (B*S, H); A-frags for all K held in registers,
// W_attn columns streamed (1 MB, cache resident). Fused tanh*v + reduction.
// ---------------------------------------------------------------------------
__global__ void __launch_bounds__(256)
attn_scores_kernel(const float* __restrict__ inter,     // (B*S, H)
                   const float* __restrict__ W_attn,    // (H, 2H)
                   const float* __restrict__ vvec,      // (H)
                   const float* __restrict__ hid_proj,  // (B, H)
                   float* __restrict__ scores) {        // (B*S)
    int lane = threadIdx.x & 31;
    int gw = blockIdx.x * (blockDim.x >> 5) + (threadIdx.x >> 5);
    if (gw >= (Bc * Sc) / 16) return;
    int m0 = gw * 16;
    int half = lane >> 4;

    v16bf afr[16];
#pragma unroll
    for (int kk = 0; kk < 16; ++kk)
        afr[kk] = load_a_f32(inter, Hc, m0, kk * 32, lane);

    float sacc[8];
#pragma unroll
    for (int r = 0; r < 8; ++r) sacc[r] = 0.0f;

    for (int nt = 0; nt < Hc / 16; ++nt) {
        int n0 = nt * 16;
        v8f acc = zero_v8f();
#pragma unroll
        for (int kk = 0; kk < 16; ++kk) {
            v16bf b = load_b_f32(W_attn, 2 * Hc, n0, kk * 32, lane);
            acc = WMMA_BF16(afr[kk], b, acc);
        }
        int n = n0 + (lane & 15);
        float vn = vvec[n];
#pragma unroll
        for (int r = 0; r < 8; ++r) {
            int grow = m0 + r + 8 * half;
            int bidx = grow / Sc;
            float e = tanhf(acc[r] + hid_proj[(long)bidx * Hc + n]);
            sacc[r] += e * vn;
        }
    }
    // reduce over n: lanes 0..15 (and 16..31) hold n = lane%16
#pragma unroll
    for (int r = 0; r < 8; ++r) {
        float s = sacc[r];
        for (int off = 1; off < 16; off <<= 1) s += __shfl_xor(s, off, 16);
        sacc[r] = s;
    }
    if ((lane & 15) == 0) {
#pragma unroll
        for (int r = 0; r < 8; ++r)
            scores[m0 + r + 8 * half] = sacc[r];
    }
}

// ---------------------------------------------------------------------------
// Softmax over S + context = attn_w @ inter_output.  One block per batch row.
// ---------------------------------------------------------------------------
__global__ void softmax_context_kernel(const float* __restrict__ scores,
                                       const float* __restrict__ inter,
                                       float* __restrict__ context) {
    int b = blockIdx.x;
    __shared__ float w[Sc];
    if (threadIdx.x == 0) {
        float mx = -1e30f;
        for (int s = 0; s < Sc; ++s) mx = fmaxf(mx, scores[b * Sc + s]);
        float sum = 0.0f;
        for (int s = 0; s < Sc; ++s) {
            float e = __expf(scores[b * Sc + s] - mx);
            w[s] = e;
            sum += e;
        }
        float inv = 1.0f / sum;
        for (int s = 0; s < Sc; ++s) w[s] *= inv;
    }
    __syncthreads();
    for (int h = threadIdx.x; h < Hc; h += blockDim.x) {
        float acc = 0.0f;
        for (int s = 0; s < Sc; ++s)
            acc += w[s] * inter[((long)b * Sc + s) * Hc + h];
        context[(long)b * Hc + h] = acc;
    }
}

// ---------------------------------------------------------------------------
// x = [emb | context] @ W_comb^T + b_comb.  A-rows gathered on the fly:
// k<H comes from emb_table[input[m]], k>=H from context[m].
// ---------------------------------------------------------------------------
__global__ void comb_kernel(const int* __restrict__ inp,
                            const float* __restrict__ emb_table,
                            const float* __restrict__ context,
                            const float* __restrict__ W_comb,
                            const float* __restrict__ b_comb,
                            float* __restrict__ x_out,  // d_out region (B,H)
                            float* __restrict__ x_ws) { // ws copy
    int lane = threadIdx.x & 31;
    int gw = blockIdx.x * (blockDim.x >> 5) + (threadIdx.x >> 5);
    if (gw >= (Bc / 16) * (Hc / 16)) return;
    int nt = gw % (Hc / 16), mt = gw / (Hc / 16);
    int m0 = mt * 16, n0 = nt * 16;
    int half = lane >> 4;

    int grow = m0 + (lane & 15);
    long erow = (long)inp[grow] * Hc;

    v8f acc = zero_v8f();
    for (int k0 = 0; k0 < 2 * Hc; k0 += 32) {
        const float* rowbase = (k0 < Hc)
                                   ? (emb_table + erow + k0)
                                   : (context + (long)grow * Hc + (k0 - Hc));
        const float4* q = (const float4*)(rowbase + half * 8);
        v16bf a = cat16(cvt8(q[0], q[1]), cvt8(q[4], q[5]));
        v16bf b = load_b_f32(W_comb, 2 * Hc, n0, k0, lane);
        acc = WMMA_BF16(a, b, acc);
    }
    int n = n0 + (lane & 15);
    float bv = b_comb[n];
#pragma unroll
    for (int r = 0; r < 8; ++r) {
        float val = acc[r] + bv;
        long o = (long)(m0 + r + 8 * half) * Hc + n;
        x_out[o] = val;
        x_ws[o] = val;
    }
}

// ---------------------------------------------------------------------------
// GRU elementwise combine.
// ---------------------------------------------------------------------------
__global__ void gru_kernel(const float* __restrict__ gi,
                           const float* __restrict__ gh,
                           const float* __restrict__ hprev,
                           float* __restrict__ hnew_out,
                           float* __restrict__ gru_out,
                           float* __restrict__ hnew_ws) {
    int i = blockIdx.x * blockDim.x + threadIdx.x;  // over B*H
    int b = i / Hc, hh = i % Hc;
    long gb = (long)b * 3 * Hc + hh;
    float ir = gi[gb], iz = gi[gb + Hc], inn = gi[gb + 2 * Hc];
    float hr = gh[gb], hz = gh[gb + Hc], hn = gh[gb + 2 * Hc];
    float r = 1.0f / (1.0f + __expf(-(ir + hr)));
    float z = 1.0f / (1.0f + __expf(-(iz + hz)));
    float n = tanhf(inn + r * hn);
    float hv = (1.0f - z) * n + z * hprev[i];
    hnew_out[i] = hv;
    gru_out[i] = hv;
    hnew_ws[i] = hv;
}

// ---------------------------------------------------------------------------
// Vocab GEMM: out(B,V) = h_new(B,H) @ W_out(V,H)^T + b_out.
// W_out streamed from HBM exactly once: each wave owns 16 vocab columns x all
// 256 batch rows (16 fp32 accumulator tiles = 128 VGPRs). h_new is staged per
// 64-wide K-panel into LDS as bf16 (padded stride 72 to avoid bank conflicts)
// and shared by all 8 waves of the block.
// ---------------------------------------------------------------------------
__global__ void __launch_bounds__(256)
vocab_gemm_kernel(const float* __restrict__ hnew,   // (B,H)
                  const float* __restrict__ W_out,  // (V,H)
                  const float* __restrict__ b_out,  // (V)
                  float* __restrict__ out) {        // (B,V)
    constexpr int LDA_P = 72;  // bf16 elements per LDS row (64 + 8 pad)
    __shared__ bf16_t Ap[256 * LDA_P];

    int lane = threadIdx.x & 31;
    int wave = threadIdx.x >> 5;
    int n0 = blockIdx.x * 128 + wave * 16;
    bool active = (n0 < Vc);  // V % 16 == 0, so whole waves go idle only
    int n = n0 + (lane & 15);
    int half = lane >> 4;

    v8f acc[16];
#pragma unroll
    for (int t = 0; t < 16; ++t) acc[t] = zero_v8f();

    for (int kp = 0; kp < Hc; kp += 64) {
        __syncthreads();
        {   // cooperative A-panel load: thread t -> row t, cols kp..kp+63
            int row = threadIdx.x;
            const float4* src = (const float4*)(hnew + (long)row * Hc + kp);
            bf16_t* dst = &Ap[row * LDA_P];
#pragma unroll
            for (int j = 0; j < 8; ++j) {
                v8bf vv = cvt8(src[2 * j], src[2 * j + 1]);
                *(v8bf*)(dst + 8 * j) = vv;
            }
        }
        __syncthreads();

        if (active) {
            // prefetch next K-panel of our W_out columns into GL2
            if (kp + 64 < Hc)
                __builtin_prefetch(W_out + (long)n * Hc + kp + 64, 0, 1);
#pragma unroll
            for (int ks = 0; ks < 64; ks += 32) {
                v16bf bf = load_b_f32(W_out, Hc, n0, kp + ks, lane);
#pragma unroll
                for (int mt = 0; mt < 16; ++mt) {
                    const bf16_t* rp =
                        &Ap[(mt * 16 + (lane & 15)) * LDA_P + ks + half * 8];
                    v8bf lo = *(const v8bf*)rp;
                    v8bf hi = *(const v8bf*)(rp + 16);
                    acc[mt] = WMMA_BF16(cat16(lo, hi), bf, acc[mt]);
                }
            }
        }
    }

    if (active) {
        float bv = b_out[n];
#pragma unroll
        for (int mt = 0; mt < 16; ++mt)
#pragma unroll
            for (int r = 0; r < 8; ++r)
                out[(long)(mt * 16 + r + 8 * half) * Vc + n] = acc[mt][r] + bv;
    }
}

// ---------------------------------------------------------------------------
// Launch
// ---------------------------------------------------------------------------
extern "C" void kernel_launch(void* const* d_in, const int* in_sizes, int n_in,
                              void* d_out, int out_size, void* d_ws, size_t ws_size,
                              hipStream_t stream) {
    const int*   inp       = (const int*)d_in[0];
    const float* hidden    = (const float*)d_in[1];   // (1,B,H)
    const float* inter     = (const float*)d_in[2];   // (B,S,H)
    const float* emb_table = (const float*)d_in[3];   // (V,H)
    const float* W_attn    = (const float*)d_in[4];   // (H,2H)
    const float* b_attn    = (const float*)d_in[5];
    const float* vvec      = (const float*)d_in[6];   // (1,H)
    const float* W_comb    = (const float*)d_in[7];   // (H,2H)
    const float* b_comb    = (const float*)d_in[8];
    const float* W_ih      = (const float*)d_in[9];   // (3H,H)
    const float* W_hh      = (const float*)d_in[10];  // (3H,H)
    const float* b_ih      = (const float*)d_in[11];
    const float* b_hh      = (const float*)d_in[12];
    const float* W_out     = (const float*)d_in[13];  // (V,H)
    const float* b_out     = (const float*)d_in[14];

    // Output regions, concatenated flat in return order.
    float* out      = (float*)d_out;               // (B,1,V)
    float* hnew_out = out + (long)Bc * Vc;         // (1,B,H)
    float* x_out    = hnew_out + (long)Bc * Hc;    // (B,1,H)
    float* gru_out  = x_out + (long)Bc * Hc;       // (B,1,H)

    // Workspace carve-up (fp32).
    float* ws       = (float*)d_ws;
    float* hid_proj = ws;                          // B*H
    float* scores   = hid_proj + (long)Bc * Hc;    // B*S
    float* context  = scores + (long)Bc * Sc;      // B*H
    float* xw       = context + (long)Bc * Hc;     // B*H
    float* gi       = xw + (long)Bc * Hc;          // B*3H
    float* gh       = gi + (long)Bc * 3 * Hc;      // B*3H
    float* hnew_ws  = gh + (long)Bc * 3 * Hc;      // B*H

    // 1) hid_proj = h @ W_attn[:,H:].T + b_attn    (16x32 tiles = 512 waves)
    gemm_bias_kernel<<<64, 256, 0, stream>>>(hidden, Hc, W_attn + Hc, 2 * Hc,
                                             b_attn, hid_proj, Hc,
                                             Bc / 16, Hc / 16, Hc);
    // 2) scores (800 tiles of 16 rows)
    attn_scores_kernel<<<100, 256, 0, stream>>>(inter, W_attn, vvec, hid_proj,
                                                scores);
    // 3) softmax + context
    softmax_context_kernel<<<Bc, 256, 0, stream>>>(scores, inter, context);
    // 4) x = [emb|context] @ W_comb^T + b_comb   (16x32 tiles = 512 waves)
    comb_kernel<<<64, 256, 0, stream>>>(inp, emb_table, context, W_comb, b_comb,
                                        x_out, xw);
    // 5) gi = x @ W_ih^T + b_ih   (16x96 tiles = 1536 waves)
    gemm_bias_kernel<<<192, 256, 0, stream>>>(xw, Hc, W_ih, Hc, b_ih, gi,
                                              3 * Hc, Bc / 16, 3 * Hc / 16, Hc);
    // 6) gh = h @ W_hh^T + b_hh
    gemm_bias_kernel<<<192, 256, 0, stream>>>(hidden, Hc, W_hh, Hc, b_hh, gh,
                                              3 * Hc, Bc / 16, 3 * Hc / 16, Hc);
    // 7) GRU combine -> h_new / gru_output
    gru_kernel<<<(Bc * Hc) / 256, 256, 0, stream>>>(gi, gh, hidden, hnew_out,
                                                    gru_out, hnew_ws);
    // 8) out = h_new @ W_out^T + b_out   (6250 N-tiles, 8 per block)
    vocab_gemm_kernel<<<(Vc / 16 + 7) / 8, 256, 0, stream>>>(hnew_ws, W_out,
                                                             b_out, out);
}
